// DCNv3_C_85160611545644
// MI455X (gfx1250) — matrix operand order repaired and checked by reading, not compiled, not run on hardware
//
#include <hip/hip_runtime.h>
#include <hip/hip_bf16.h>
#include <math.h>

// ---------------------------------------------------------------------------
// DCNv3_C for gfx1250 (MI455X, wave32, WMMA).
//   K1: y   = conv1x1(x)            GEMM 32768x256x192  (A strided from NCHW)
//   K2: val = y @ in_w^T            GEMM 32768x256x256
//   K3: t   = gelu(LN(dwconv(y)))   per-position block, LDS reduction
//   K4: off = t @ off_w^T (N=72), msk = t @ mask_w^T (N=36)
//   K5: acc = deform-sample(val, off, softmax(msk))
//   K6: out = acc @ out_w^T -> NCHW scatter
// GEMM: v_wmma_f32_16x16x32_bf16, fp32 accumulate, 32x32 wave tile,
// double-buffered LDS with register-staged global prefetch.
// ---------------------------------------------------------------------------

typedef __bf16 bf16_t;
typedef bf16_t v16bf __attribute__((ext_vector_type(16)));
typedef bf16_t v8bf  __attribute__((ext_vector_type(8)));
typedef float  v8f   __attribute__((ext_vector_type(8)));

#define TILE_M 128
#define TILE_N 64
#define TILE_K 32

// ---------------------------------------------------------------------------
// Generic bf16-WMMA GEMM:  C[m,n] = sum_k A[m,k] * W[n,k] + bias[n]
//   a_nchw   : A element (m,k) at A[(m>>12)*(K<<12) + (k<<12) + (m&4095)]
//              (NCHW tensor with H*W = 4096), else row-major lda = K.
//   out_nchw : scatter C to NCHW [Nreal channels, 4096 spatial], else ldc.
//   Nreal may be ragged (72/36): B rows >= Nreal are zero, stores guarded.
// Block: 256 threads = 8 waves (4 in M x 2 in N); wave tile 32x32 (4 WMMA).
// ---------------------------------------------------------------------------
__global__ __launch_bounds__(256)
void gemm_bf16_wmma(const float* __restrict__ A, const float* __restrict__ W,
                    const float* __restrict__ bias, float* __restrict__ Cout,
                    int M, int Nreal, int K, int ldc, int a_nchw, int out_nchw)
{
    __shared__ bf16_t sA[2][TILE_M * TILE_K];   // [m][k], k contiguous
    __shared__ bf16_t sB[2][TILE_N * TILE_K];   // transposed: [n][k], k contiguous

    const int tid  = threadIdx.x;
    const int lane = tid & 31;
    const int wave = tid >> 5;
    const int wm   = (wave & 3) * 32;        // wave M offset in tile
    const int wn   = (wave >> 2) * 32;       // wave N offset in tile

    const int tileM = blockIdx.x * TILE_M;
    const int tileN = blockIdx.y * TILE_N;

    // staging assignment: A -> 16 k-elements/thread, B -> 8 k-elements/thread
    const int rA  = tid >> 1;                // 0..127
    const int kcA = (tid & 1) * 16;
    const int rB  = tid >> 2;                // 0..63
    const int kcB = (tid & 3) * 8;

    const int nk = K / TILE_K;
    float aR[16];
    float bR[8];

    // ---- register-stage K-tile kt ----
    auto load_regs = [&](int kt) {
        const int k0 = kt * TILE_K;
        const int m  = tileM + rA;
        if (a_nchw) {
            const float* src = A + (size_t)(m >> 12) * ((size_t)K << 12) + (m & 4095);
            #pragma unroll
            for (int j = 0; j < 16; ++j)
                aR[j] = src[(size_t)(k0 + kcA + j) << 12];
        } else {
            const float* src = A + (size_t)m * K + k0 + kcA;
            #pragma unroll
            for (int j = 0; j < 16; ++j) aR[j] = src[j];
        }
        const int n = tileN + rB;
        if (n < Nreal) {
            const float* src = W + (size_t)n * K + k0 + kcB;
            #pragma unroll
            for (int j = 0; j < 8; ++j) bR[j] = src[j];
        } else {
            #pragma unroll
            for (int j = 0; j < 8; ++j) bR[j] = 0.f;
        }
    };
    auto store_lds = [&](int buf) {
        #pragma unroll
        for (int j = 0; j < 16; ++j)
            sA[buf][rA * TILE_K + kcA + j] = (bf16_t)aR[j];
        #pragma unroll
        for (int j = 0; j < 8; ++j)
            sB[buf][rB * TILE_K + kcB + j] = (bf16_t)bR[j];
    };

    v8f acc00 = {}, acc01 = {}, acc10 = {}, acc11 = {};

    load_regs(0);
    store_lds(0);
    __syncthreads();

    for (int kt = 0; kt < nk; ++kt) {
        const int buf = kt & 1;
        if (kt + 1 < nk) load_regs(kt + 1);   // overlaps with WMMAs below

        // ---- fragments per ISA VGPR layouts ----
        // A 16x32 bf16: lanes 0-15 -> M=lane,    K = 0..7  & 16..23
        //               lanes 16-31 -> M=lane-16, K = 8..15 & 24..31
        const int mf  = lane & 15;
        const int kbA = (lane < 16) ? 0 : 8;
        union { v16bf v; v8bf h[2]; } a0, a1, b0, b1;
        a0.h[0] = *(const v8bf*)&sA[buf][(wm + mf) * TILE_K + kbA];
        a0.h[1] = *(const v8bf*)&sA[buf][(wm + mf) * TILE_K + kbA + 16];
        a1.h[0] = *(const v8bf*)&sA[buf][(wm + 16 + mf) * TILE_K + kbA];
        a1.h[1] = *(const v8bf*)&sA[buf][(wm + 16 + mf) * TILE_K + kbA + 16];
        // B 32x16 bf16: lanes 0-15 -> N=lane, K=0..7 & 16..23 ; lanes 16-31 -> K=8..15 & 24..31
        const int kbB = (lane < 16) ? 0 : 8;
        b0.h[0] = *(const v8bf*)&sB[buf][(wn + mf) * TILE_K + kbB];
        b0.h[1] = *(const v8bf*)&sB[buf][(wn + mf) * TILE_K + kbB + 16];
        b1.h[0] = *(const v8bf*)&sB[buf][(wn + 16 + mf) * TILE_K + kbB];
        b1.h[1] = *(const v8bf*)&sB[buf][(wn + 16 + mf) * TILE_K + kbB + 16];

        acc00 = __builtin_amdgcn_wmma_f32_16x16x32_bf16(
                    false, a0.v, false, b0.v, (short)0, acc00, false, false);
        acc01 = __builtin_amdgcn_wmma_f32_16x16x32_bf16(
                    false, a0.v, false, b1.v, (short)0, acc01, false, false);
        acc10 = __builtin_amdgcn_wmma_f32_16x16x32_bf16(
                    false, a1.v, false, b0.v, (short)0, acc10, false, false);
        acc11 = __builtin_amdgcn_wmma_f32_16x16x32_bf16(
                    false, a1.v, false, b1.v, (short)0, acc11, false, false);

        if (kt + 1 < nk) {
            store_lds(buf ^ 1);               // other buffer: no WAR on current reads
            __syncthreads();
        }
    }

    // ---- epilogue: C/D layout: VGPR rr -> M = rr + 8*(lane>=16), N = lane&15 ----
    const int mrow0 = tileM + wm + ((lane < 16) ? 0 : 8);
    const int nc0   = tileN + wn + (lane & 15);
    const int nc1   = nc0 + 16;
    const float bb0 = (nc0 < Nreal) ? bias[nc0] : 0.f;
    const float bb1 = (nc1 < Nreal) ? bias[nc1] : 0.f;
    #pragma unroll
    for (int rr = 0; rr < 8; ++rr) {
        const int mA = mrow0 + rr;            // rows of acc0x
        const int mB = mA + 16;               // rows of acc1x
        if (out_nchw) {
            const size_t baseA = (size_t)(mA >> 12) * ((size_t)Nreal << 12) + (mA & 4095);
            const size_t baseB = (size_t)(mB >> 12) * ((size_t)Nreal << 12) + (mB & 4095);
            if (nc0 < Nreal) {
                Cout[baseA + ((size_t)nc0 << 12)] = acc00[rr] + bb0;
                Cout[baseB + ((size_t)nc0 << 12)] = acc10[rr] + bb0;
            }
            if (nc1 < Nreal) {
                Cout[baseA + ((size_t)nc1 << 12)] = acc01[rr] + bb1;
                Cout[baseB + ((size_t)nc1 << 12)] = acc11[rr] + bb1;
            }
        } else {
            const size_t baseA = (size_t)mA * ldc;
            const size_t baseB = (size_t)mB * ldc;
            if (nc0 < Nreal) {
                Cout[baseA + nc0] = acc00[rr] + bb0;
                Cout[baseB + nc0] = acc10[rr] + bb0;
            }
            if (nc1 < Nreal) {
                Cout[baseA + nc1] = acc01[rr] + bb1;
                Cout[baseB + nc1] = acc11[rr] + bb1;
            }
        }
    }
}

// ---------------------------------------------------------------------------
// K3: depthwise 3x3 (zero pad) + LayerNorm(C=256) + exact GELU.
// One 256-thread block per spatial position; thread = channel.
// ---------------------------------------------------------------------------
__global__ __launch_bounds__(256)
void dwconv_ln_gelu(const float* __restrict__ y, const float* __restrict__ dw_w,
                    const float* __restrict__ dw_b, const float* __restrict__ ln_g,
                    const float* __restrict__ ln_b, float* __restrict__ t)
{
    const int m  = blockIdx.x;           // n*4096 + h*64 + w
    const int c  = threadIdx.x;
    const int n  = m >> 12;
    const int hw = m & 4095;
    const int h  = hw >> 6, w = hw & 63;

    float v = dw_b[c];
    #pragma unroll
    for (int i = 0; i < 3; ++i) {
        const int yh = h + i - 1;
        if (yh < 0 || yh > 63) continue;
        #pragma unroll
        for (int j = 0; j < 3; ++j) {
            const int yw = w + j - 1;
            if (yw < 0 || yw > 63) continue;
            v += y[(size_t)((((n << 6) + yh) << 6) + yw) * 256 + c] * dw_w[c * 9 + i * 3 + j];
        }
    }

    __shared__ float s1[256], s2[256];
    s1[c] = v; s2[c] = v * v;
    __syncthreads();
    for (int off = 128; off > 0; off >>= 1) {
        if (c < off) { s1[c] += s1[c + off]; s2[c] += s2[c + off]; }
        __syncthreads();
    }
    const float mean = s1[0] * (1.f / 256.f);
    const float var  = s2[0] * (1.f / 256.f) - mean * mean;
    const float xn   = (v - mean) * rsqrtf(var + 1e-5f) * ln_g[c] + ln_b[c];
    const float ge   = 0.5f * xn * (1.f + erff(xn * 0.70710678118654752f));
    t[(size_t)m * 256 + c] = ge;
}

// ---------------------------------------------------------------------------
// K5: deformable sampling. One wave per (position m, group g); 32 lanes x 2
// channels cover the 64 group channels. ref+grid+offset algebra collapses to:
//   ix = w + p/3 + off_x,  iy = h + p%3 + off_y   (padded 66x66 coords)
// pad ring is zero, so contributions only when 1 <= c{x,y} <= 64.
// ---------------------------------------------------------------------------
__global__ __launch_bounds__(256)
void dcn_sample(const float* __restrict__ val, const float* __restrict__ offb,
                const float* __restrict__ mskb, float* __restrict__ accb)
{
    const int gwave = blockIdx.x * 8 + (threadIdx.x >> 5);
    const int lane  = threadIdx.x & 31;
    const int m = gwave >> 2;
    const int g = gwave & 3;
    const int n  = m >> 12;
    const int hw = m & 4095;
    const int h  = hw >> 6, w = hw & 63;

    const float* off = offb + (size_t)m * 72 + g * 18;
    const float* msk = mskb + (size_t)m * 36 + g * 9;

    // softmax over the 9 mask logits (uniform across the wave -> broadcast loads)
    float mk[9];
    float mmax = -1e30f;
    #pragma unroll
    for (int p = 0; p < 9; ++p) { mk[p] = msk[p]; mmax = fmaxf(mmax, mk[p]); }
    float msum = 0.f;
    #pragma unroll
    for (int p = 0; p < 9; ++p) { mk[p] = __expf(mk[p] - mmax); msum += mk[p]; }
    const float minv = 1.f / msum;

    const int cbase = g * 64 + lane * 2;
    float2 acc = make_float2(0.f, 0.f);
    #pragma unroll
    for (int p = 0; p < 9; ++p) {
        const float ix = (float)(w + p / 3) + off[2 * p];
        const float iy = (float)(h + p % 3) + off[2 * p + 1];
        const float x0 = floorf(ix), y0 = floorf(iy);
        const float fx = ix - x0, fy = iy - y0;
        const int xi = (int)x0, yi = (int)y0;
        const float wp = mk[p] * minv;
        #pragma unroll
        for (int b = 0; b < 2; ++b) {
            const int cy = yi + b;
            if (cy < 1 || cy > 64) continue;
            const float wy = b ? fy : (1.f - fy);
            #pragma unroll
            for (int a = 0; a < 2; ++a) {
                const int cx = xi + a;
                if (cx < 1 || cx > 64) continue;
                const float wx = a ? fx : (1.f - fx);
                const float2 vv = *(const float2*)&val[
                    (size_t)((((n << 6) + (cy - 1)) << 6) + (cx - 1)) * 256 + cbase];
                const float wgt = wp * wx * wy;
                acc.x += wgt * vv.x;
                acc.y += wgt * vv.y;
            }
        }
    }
    *(float2*)&accb[(size_t)m * 256 + cbase] = acc;
}

// ---------------------------------------------------------------------------
extern "C" void kernel_launch(void* const* d_in, const int* in_sizes, int n_in,
                              void* d_out, int out_size, void* d_ws, size_t ws_size,
                              hipStream_t stream)
{
    (void)in_sizes; (void)n_in; (void)out_size; (void)ws_size;

    const float* x      = (const float*)d_in[0];
    const float* conv_w = (const float*)d_in[1];
    const float* conv_b = (const float*)d_in[2];
    const float* dw_w   = (const float*)d_in[3];
    const float* dw_b   = (const float*)d_in[4];
    const float* ln_g   = (const float*)d_in[5];
    const float* ln_b   = (const float*)d_in[6];
    const float* in_w   = (const float*)d_in[7];
    const float* in_b   = (const float*)d_in[8];
    const float* off_w  = (const float*)d_in[9];
    const float* off_b  = (const float*)d_in[10];
    const float* mask_w = (const float*)d_in[11];
    const float* mask_b = (const float*)d_in[12];
    const float* out_w  = (const float*)d_in[13];
    const float* out_b  = (const float*)d_in[14];

    // workspace layout (fp32 elements); total ~115 MB — fits in L2 (192 MB).
    const size_t MHW = 32768;
    float* ws   = (float*)d_ws;
    float* y    = ws;                        // [32768,256] NHWC; reused as acc
    float* val  = ws + MHW * 256;            // [32768,256]
    float* t    = ws + 2 * MHW * 256;        // [32768,256]
    float* offb = ws + 3 * MHW * 256;        // [32768,72]
    float* mskb = offb + MHW * 72;           // [32768,36]
    float* acc  = y;                         // y dead after K3

    const dim3 blk(256);
    // K1: y = conv1x1(x) + b     (A strided out of NCHW)
    gemm_bf16_wmma<<<dim3(256, 4), blk, 0, stream>>>(x, conv_w, conv_b, y,
                                                     32768, 256, 192, 256, 1, 0);
    // K2: val = y @ in_w^T + in_b
    gemm_bf16_wmma<<<dim3(256, 4), blk, 0, stream>>>(y, in_w, in_b, val,
                                                     32768, 256, 256, 256, 0, 0);
    // K3: t = gelu(LN(dwconv(y)))
    dwconv_ln_gelu<<<32768, blk, 0, stream>>>(y, dw_w, dw_b, ln_g, ln_b, t);
    // K4a: offsets (ragged N=72)
    gemm_bf16_wmma<<<dim3(256, 2), blk, 0, stream>>>(t, off_w, off_b, offb,
                                                     32768, 72, 256, 72, 0, 0);
    // K4b: mask logits (ragged N=36)
    gemm_bf16_wmma<<<dim3(256, 1), blk, 0, stream>>>(t, mask_w, mask_b, mskb,
                                                     32768, 36, 256, 36, 0, 0);
    // K5: deformable bilinear sampling (131072 waves)
    dcn_sample<<<16384, blk, 0, stream>>>(val, offb, mskb, acc);
    // K6: out = acc @ out_w^T + out_b, scattered to NCHW
    gemm_bf16_wmma<<<dim3(256, 4), blk, 0, stream>>>(acc, out_w, out_b, (float*)d_out,
                                                     32768, 256, 256, 256, 0, 1);
}
